// Attention_7739531068052
// MI455X (gfx1250) — matrix operand order repaired and checked by reading, not compiled, not run on hardware
//
#include <hip/hip_runtime.h>

typedef __attribute__((ext_vector_type(8)))  __bf16 bf16x8;
typedef __attribute__((ext_vector_type(16))) __bf16 bf16x16;
typedef __attribute__((ext_vector_type(8)))  float  f32x8;

#define HEADS 12
#define NW    49
#define HD    32
#define CDIM  384
#define NROWS 200704   /* 4096*49 */

static __device__ __forceinline__ bf16x16 cat8(bf16x8 lo, bf16x8 hi) {
  bf16x16 r;
#pragma unroll
  for (int i = 0; i < 8; ++i) { r[i] = lo[i]; r[i + 8] = hi[i]; }
  return r;
}

static __device__ __forceinline__ bf16x16 zero_frag() {
  bf16x16 r;
#pragma unroll
  for (int i = 0; i < 16; ++i) r[i] = (__bf16)0.0f;
  return r;
}

static __device__ __forceinline__ f32x8 wmma_bf16(bf16x16 a, bf16x16 b, f32x8 c) {
  return __builtin_amdgcn_wmma_f32_16x16x32_bf16(false, a, false, b, (short)0, c,
                                                 false, false);
}

// ---------------------------------------------------------------------------
// Tiled GEMM: Out[m,n] = sum_k A[m,k] * W[n,k] + bias[n]
// Block tile 128(M) x 64(N), K-step 32, 8 waves, each wave owns 32x32.
// ---------------------------------------------------------------------------
template <int A_BF16, int OUT_F32>
__global__ __launch_bounds__(256) void gemm_wmma_kernel(
    const void* __restrict__ Av, const float* __restrict__ W,
    const float* __restrict__ bias, void* __restrict__ Outv, int N, int K) {
  __shared__ __attribute__((aligned(16))) __bf16 aTile[128 * 32];  // 8 KB
  __shared__ __attribute__((aligned(16))) __bf16 wTile[64 * 32];   // 4 KB

  const int t     = threadIdx.x;
  const int lane  = t & 31;
  const int ln15  = lane & 15;
  const int wid   = t >> 5;
  const int waveM = wid & 3;   // 0..3  -> 32-row slab
  const int waveN = wid >> 2;  // 0..1  -> 32-col slab
  const long m0   = (long)blockIdx.x * 128;
  const int  n0   = blockIdx.y * 64;

  // staging assignments
  const int arow = t >> 1;        // 0..127
  const int acg  = (t & 1) * 16;  // 0 / 16
  const int wrow = t >> 2;        // 0..63
  const int wcg  = (t & 3) * 8;   // 0,8,16,24

  f32x8 acc[2][2] = {};

  for (int k0 = 0; k0 < K; k0 += 32) {
    // prefetch next K-tile while this one is staged/consumed
    if (k0 + 32 < K) {
      if (A_BF16)
        __builtin_prefetch((const __bf16*)Av + (m0 + arow) * (long)K + k0 + 32 + acg, 0, 1);
      else
        __builtin_prefetch((const float*)Av + (m0 + arow) * (long)K + k0 + 32 + acg, 0, 1);
      __builtin_prefetch(W + (long)(n0 + wrow) * K + k0 + 32 + wcg, 0, 1);
    }
    __syncthreads();
    // ---- stage A tile (convert f32 -> bf16 if needed) ----
    if (A_BF16) {
      const __bf16* A = (const __bf16*)Av;
      const bf16x8* src =
          (const bf16x8*)(A + (m0 + arow) * (long)K + k0 + acg);
      *(bf16x8*)&aTile[arow * 32 + acg]     = src[0];
      *(bf16x8*)&aTile[arow * 32 + acg + 8] = src[1];
    } else {
      const float* A = (const float*)Av;
      const float4* src =
          (const float4*)(A + (m0 + arow) * (long)K + k0 + acg);
      float4 f0 = src[0], f1 = src[1], f2 = src[2], f3 = src[3];
      bf16x8 lo, hi;
      lo[0] = (__bf16)f0.x; lo[1] = (__bf16)f0.y; lo[2] = (__bf16)f0.z; lo[3] = (__bf16)f0.w;
      lo[4] = (__bf16)f1.x; lo[5] = (__bf16)f1.y; lo[6] = (__bf16)f1.z; lo[7] = (__bf16)f1.w;
      hi[0] = (__bf16)f2.x; hi[1] = (__bf16)f2.y; hi[2] = (__bf16)f2.z; hi[3] = (__bf16)f2.w;
      hi[4] = (__bf16)f3.x; hi[5] = (__bf16)f3.y; hi[6] = (__bf16)f3.z; hi[7] = (__bf16)f3.w;
      *(bf16x8*)&aTile[arow * 32 + acg]     = lo;
      *(bf16x8*)&aTile[arow * 32 + acg + 8] = hi;
    }
    // ---- stage W tile (f32 -> bf16) ----
    {
      const float4* src =
          (const float4*)(W + (long)(n0 + wrow) * K + k0 + wcg);
      float4 g0 = src[0], g1 = src[1];
      bf16x8 wv;
      wv[0] = (__bf16)g0.x; wv[1] = (__bf16)g0.y; wv[2] = (__bf16)g0.z; wv[3] = (__bf16)g0.w;
      wv[4] = (__bf16)g1.x; wv[5] = (__bf16)g1.y; wv[6] = (__bf16)g1.z; wv[7] = (__bf16)g1.w;
      *(bf16x8*)&wTile[wrow * 32 + wcg] = wv;
    }
    __syncthreads();

    // ---- fragments + WMMA ----
    bf16x16 af[2], wf[2];
    const int acb = (lane >> 4) * 8;    // A interleave: K=[cb,cb+8)+[cb+16,cb+24)
    const int wkb = (lane >> 4) * 8;    // B interleave: K=[kb,kb+8)+[kb+16,kb+24)
#pragma unroll
    for (int i = 0; i < 2; ++i) {
      const int row = waveM * 32 + i * 16 + ln15;
      af[i] = cat8(*(const bf16x8*)&aTile[row * 32 + acb],
                   *(const bf16x8*)&aTile[row * 32 + acb + 16]);
    }
#pragma unroll
    for (int j = 0; j < 2; ++j) {
      const int row = waveN * 32 + j * 16 + ln15;
      wf[j] = cat8(*(const bf16x8*)&wTile[row * 32 + wkb],
                   *(const bf16x8*)&wTile[row * 32 + wkb + 16]);
    }
#pragma unroll
    for (int i = 0; i < 2; ++i)
#pragma unroll
      for (int j = 0; j < 2; ++j) acc[i][j] = wmma_bf16(af[i], wf[j], acc[i][j]);
  }

  // ---- epilogue: + bias, store ----
  const int hi8 = (lane >> 4) * 8;
#pragma unroll
  for (int i = 0; i < 2; ++i) {
#pragma unroll
    for (int j = 0; j < 2; ++j) {
      const long mb = m0 + waveM * 32 + i * 16 + hi8;
      const int  gn = n0 + waveN * 32 + j * 16 + ln15;
      const float bn = bias[gn];
#pragma unroll
      for (int r = 0; r < 8; ++r) {
        const long  m = mb + r;
        const float v = acc[i][j][r] + bn;
        if (OUT_F32)
          ((float*)Outv)[m * (long)N + gn] = v;
        else
          ((__bf16*)Outv)[m * (long)N + gn] = (__bf16)v;
      }
    }
  }
}

// ---------------------------------------------------------------------------
// Fused attention per (window b, head h). One wave per block.
// Bias is staged transposed+padded (biasT[kk][q], 64x64, -1e30 sentinels) so
// the scale step is a branch-free fmaf with vector ds_load_b128 reads.
// ---------------------------------------------------------------------------
__global__ __launch_bounds__(32) void attn_kernel(
    const __bf16* q_ws, const __bf16* kv_ws, __bf16* o_ws,
    const float* __restrict__ rel_table, const int* __restrict__ rel_index) {
  __shared__ __attribute__((aligned(16))) float  biasT[64 * 64];  // 16 KB
  __shared__ __attribute__((aligned(16))) __bf16 pLds[64 * 64];   // 8 KB
  __shared__ __attribute__((aligned(16))) __bf16 vtLds[32 * 64];  // 4 KB

  const int b    = blockIdx.x;
  const int h    = blockIdx.y;
  const int lane = threadIdx.x;
  const int ln15 = lane & 15;
  const int hi8  = (lane >> 4) * 8;
  const int acb  = (lane >> 4) * 8;   // A-operand interleaved chunk base
  const int bkb  = (lane >> 4) * 8;   // B-operand interleaved chunk base

  // bias -> LDS, transposed [kk][q], padded with -1e30 (mask falls out of fmaf)
  for (int idx = lane; idx < 64 * 64; idx += 32) {
    const int kk = idx >> 6;
    const int q  = idx & 63;
    float v = -1e30f;
    if (kk < NW && q < NW) v = rel_table[rel_index[q * NW + kk] * HEADS + h];
    biasT[idx] = v;
  }

  // ---- Q (A-operand) and K (B-operand) fragments straight from global ----
  bf16x16 qf[4], kf[4];
#pragma unroll
  for (int i = 0; i < 4; ++i) {
    const bool valid = (i < 3) || (ln15 == 0);  // rows 48..63: only row 48 real
    if (valid) {
      const int qr = i * 16 + ln15;
      const __bf16* p = q_ws + (long)(b * NW + qr) * CDIM + h * HD + acb;
      qf[i] = cat8(*(const bf16x8*)p, *(const bf16x8*)(p + 16));
    } else {
      qf[i] = zero_frag();
    }
  }
#pragma unroll
  for (int j = 0; j < 4; ++j) {
    const bool valid = (j < 3) || (ln15 == 0);
    if (valid) {
      const int kr = j * 16 + ln15;
      const __bf16* p = kv_ws + (long)(b * NW + kr) * (2 * CDIM) + h * HD + bkb;
      kf[j] = cat8(*(const bf16x8*)p, *(const bf16x8*)(p + 16));
    } else {
      kf[j] = zero_frag();
    }
  }

  // ---- S = Q K^T (16 WMMAs) ----
  f32x8 s[4][4] = {};
#pragma unroll
  for (int i = 0; i < 4; ++i)
#pragma unroll
    for (int j = 0; j < 4; ++j) s[i][j] = wmma_bf16(qf[i], kf[j], s[i][j]);

  // ---- scale + bias + mask: branch-free, vectorized bias reads ----
  const float scale = 0.17677669529663687f;  // 32^-0.5
#pragma unroll
  for (int i = 0; i < 4; ++i)
#pragma unroll
    for (int j = 0; j < 4; ++j) {
      const int kk = j * 16 + ln15;
      const float4* bp = (const float4*)&biasT[kk * 64 + i * 16 + hi8];
      const float4 b0 = bp[0], b1 = bp[1];
      const float bb[8] = {b0.x, b0.y, b0.z, b0.w, b1.x, b1.y, b1.z, b1.w};
#pragma unroll
      for (int r = 0; r < 8; ++r) s[i][j][r] = fmaf(s[i][j][r], scale, bb[r]);
    }

  // ---- row softmax (rows live in 16-lane half-wave groups) ----
#pragma unroll
  for (int i = 0; i < 4; ++i) {
#pragma unroll
    for (int r = 0; r < 8; ++r) {
      float m = s[i][0][r];
#pragma unroll
      for (int j = 1; j < 4; ++j) m = fmaxf(m, s[i][j][r]);
#pragma unroll
      for (int off = 8; off >= 1; off >>= 1) m = fmaxf(m, __shfl_xor(m, off, 16));
      float sum = 0.0f;
#pragma unroll
      for (int j = 0; j < 4; ++j) {
        const float e = __expf(s[i][j][r] - m);
        s[i][j][r] = e;
        sum += e;
      }
#pragma unroll
      for (int off = 8; off >= 1; off >>= 1) sum += __shfl_xor(sum, off, 16);
      const float inv = 1.0f / sum;
#pragma unroll
      for (int j = 0; j < 4; ++j) s[i][j][r] *= inv;
    }
  }

  // ---- P -> LDS (bf16, row-major 64x64; 16 lanes write contiguous kk) ----
#pragma unroll
  for (int i = 0; i < 4; ++i)
#pragma unroll
    for (int j = 0; j < 4; ++j)
#pragma unroll
      for (int r = 0; r < 8; ++r) {
        const int q  = i * 16 + r + hi8;
        const int kk = j * 16 + ln15;
        pLds[q * 64 + kk] = (__bf16)s[i][j][r];
      }

  // ---- V^T -> LDS: vt[d][kk] so PV B-fragments are contiguous ----
  for (int tIdx = lane; tIdx < 64 * 32; tIdx += 32) {
    const int kk = tIdx >> 5;
    const int d  = tIdx & 31;
    __bf16 v = (__bf16)0.0f;
    if (kk < NW)
      v = kv_ws[(long)(b * NW + kk) * (2 * CDIM) + CDIM + h * HD + d];
    vtLds[d * 64 + kk] = v;
  }

  // ---- O = P V (padded K=64 -> 2 K-steps, 16 WMMAs) ----
  f32x8 o[4][2] = {};
#pragma unroll
  for (int ks = 0; ks < 2; ++ks) {
    bf16x16 pf[4], vf[2];
#pragma unroll
    for (int i = 0; i < 4; ++i) {
      const int row = i * 16 + ln15;
      const int c   = ks * 32 + acb;
      pf[i] = cat8(*(const bf16x8*)&pLds[row * 64 + c],
                   *(const bf16x8*)&pLds[row * 64 + c + 16]);
    }
#pragma unroll
    for (int jn = 0; jn < 2; ++jn) {
      const int d = jn * 16 + ln15;
      const int c = ks * 32 + bkb;
      vf[jn] = cat8(*(const bf16x8*)&vtLds[d * 64 + c],
                    *(const bf16x8*)&vtLds[d * 64 + c + 16]);
    }
#pragma unroll
    for (int i = 0; i < 4; ++i)
#pragma unroll
      for (int jn = 0; jn < 2; ++jn) o[i][jn] = wmma_bf16(pf[i], vf[jn], o[i][jn]);
  }

  // ---- store O (bf16); only tile 3 row 0 half-wave needs a predicate ----
#pragma unroll
  for (int i = 0; i < 4; ++i)
#pragma unroll
    for (int jn = 0; jn < 2; ++jn)
#pragma unroll
      for (int r = 0; r < 8; ++r) {
        const bool valid = (i < 3) || (r == 0 && hi8 == 0);  // q<49
        if (valid) {
          const int q = i * 16 + r + hi8;
          const int d = jn * 16 + ln15;
          o_ws[(long)(b * NW + q) * CDIM + h * HD + d] = (__bf16)o[i][jn][r];
        }
      }
}

// ---------------------------------------------------------------------------
extern "C" void kernel_launch(void* const* d_in, const int* in_sizes, int n_in,
                              void* d_out, int out_size, void* d_ws,
                              size_t ws_size, hipStream_t stream) {
  const float* x         = (const float*)d_in[0];
  const float* y         = (const float*)d_in[1];
  const float* q_w       = (const float*)d_in[2];
  const float* q_b       = (const float*)d_in[3];
  const float* kv_w      = (const float*)d_in[4];
  const float* kv_b      = (const float*)d_in[5];
  const float* proj_w    = (const float*)d_in[6];
  const float* proj_b    = (const float*)d_in[7];
  const float* rel_table = (const float*)d_in[8];
  const int*   rel_index = (const int*)d_in[9];
  float* out = (float*)d_out;

  __bf16* q_ws  = (__bf16*)d_ws;                 // NROWS x 384 bf16
  __bf16* kv_ws = q_ws + (size_t)NROWS * CDIM;   // NROWS x 768 bf16

  const dim3 blk(256);
  // Q = x @ q_w^T + q_b  (bf16 out)
  gemm_wmma_kernel<0, 0><<<dim3(NROWS / 128, CDIM / 64), blk, 0, stream>>>(
      (const void*)x, q_w, q_b, (void*)q_ws, CDIM, CDIM);
  // KV = y @ kv_w^T + kv_b  (bf16 out; k = cols 0..383, v = 384..767)
  gemm_wmma_kernel<0, 0><<<dim3(NROWS / 128, (2 * CDIM) / 64), blk, 0, stream>>>(
      (const void*)y, kv_w, kv_b, (void*)kv_ws, 2 * CDIM, CDIM);
  // Fused attention; output overwrites q_ws (q fully consumed per block first)
  attn_kernel<<<dim3(4096, HEADS), dim3(32), 0, stream>>>(q_ws, kv_ws, q_ws,
                                                          rel_table, rel_index);
  // out = attn_out @ proj_w^T + proj_b  (f32 out)
  gemm_wmma_kernel<1, 1><<<dim3(NROWS / 128, CDIM / 64), blk, 0, stream>>>(
      (const void*)q_ws, proj_w, proj_b, (void*)out, CDIM, CDIM);
}
